// Block_24249385353287
// MI455X (gfx1250) — matrix operand order repaired and checked by reading, not compiled, not run on hardware
//
#include <hip/hip_runtime.h>

// ---------------------------------------------------------------------------
// Types for CDNA5 WMMA (wave32): 16x16x32 bf16 -> f32 accumulate
// ---------------------------------------------------------------------------
typedef __attribute__((ext_vector_type(16))) __bf16 v16bf;
typedef __attribute__((ext_vector_type(8)))  __bf16 v8bf;
typedef __attribute__((ext_vector_type(8)))  float  v8f;

#define WMMA_BF16(a, b, c) \
  __builtin_amdgcn_wmma_f32_16x16x32_bf16(false, (a), false, (b), (short)0, (c), false, false)

// ---------------------------------------------------------------------------
// Problem constants
// ---------------------------------------------------------------------------
constexpr int Vd  = 32;
constexpr int Ed  = 64;
constexpr int Sd  = 96;
constexpr int Bn  = 128;
constexpr int CES = Vd * Ed * Sd;   // 196608 elems per batch sample
constexpr int CH  = 128;            // channels per grouped-conv block
constexpr int NG  = 24;             // groups used (S/WS)
constexpr int BLK = CH * Ed;        // 8192 elems per (batch,group) tile

// ---------------------------------------------------------------------------
// Swizzled-panel conventions (all bf16):
//   B-panel  [ntile][kchunk][frag_lane 0..31][half 0..15]
//     frag_lane = (col & 15) + 16*((kloc >> 4) & 1),  half = kloc & 15
//   A-panel  [mtile][kchunk][frag_lane 0..31][half 0..15]
//     frag_lane = (row & 15) + 16*((kloc >> 4) & 1)
//     half      = kloc & 15
// A fragment / B fragment load = one 32B contiguous per-lane LDS read.
// ---------------------------------------------------------------------------
__device__ inline v16bf frag_B_sw(const __bf16* P, int nkch, int ntile, int kchunk) {
  const int lane = threadIdx.x & 31;
  return *(const v16bf*)(P + (((ntile * nkch + kchunk) * 32) + lane) * 16);
}

__device__ inline v16bf frag_A_sw(const __bf16* P, int nkch, int mtile, int kchunk) {
  const int lane = threadIdx.x & 31;
  return *(const v16bf*)(P + (((mtile * nkch + kchunk) * 32) + lane) * 16);
}

// Scattered global load of an A fragment (used once per WG for IVGC weights).
__device__ inline v16bf frag_A_gl(const float* A, int K, int mtile, int kchunk) {
  const int lane = threadIdx.x & 31;
  const int row  = mtile * 16 + (lane & 15);
  const int kb   = kchunk * 32 + ((lane >> 4) << 4);
  v16bf a;
#pragma unroll
  for (int h = 0; h < 16; ++h)
    a[h] = (__bf16)A[(long long)row * K + kb + h];
  return a;
}

// ---------------------------------------------------------------------------
// Kernel 1: TLDC — circular depthwise conv (k=9) + dilated (k=11, dil=5).
// One wave per (b, c) row of 96; 8 rows per 256-thread block; rows in LDS.
// ---------------------------------------------------------------------------
__global__ __launch_bounds__(256)
void tldc_kernel(const float* __restrict__ x,
                 const float* __restrict__ dw1, const float* __restrict__ db1,
                 const float* __restrict__ dw2, const float* __restrict__ db2,
                 float* __restrict__ t) {
  __shared__ float srow[8][96];
  __shared__ float sloc[8][96];
  const int lr   = threadIdx.x >> 5;
  const int lane = threadIdx.x & 31;
  const int R    = blockIdx.x * 8 + lr;     // row id: b*2048 + c
  const int c    = R & 2047;
  const int b    = R >> 11;
  const float* xr = x + (long long)b * CES + (long long)c * Sd;

  float w1[9], w2[11];
#pragma unroll
  for (int j = 0; j < 9; ++j)  w1[j] = dw1[c * 9 + j];
#pragma unroll
  for (int j = 0; j < 11; ++j) w2[j] = dw2[c * 11 + j];
  const float b1 = db1[c], b2 = db2[c];

#pragma unroll
  for (int p = 0; p < 3; ++p) srow[lr][lane + p * 32] = xr[lane + p * 32];
  __syncthreads();

#pragma unroll
  for (int p = 0; p < 3; ++p) {
    const int s = lane + p * 32;
    float acc = b1;
#pragma unroll
    for (int j = 0; j < 9; ++j) {
      int sp = s + j - 4;
      sp = (sp < 0) ? sp + Sd : (sp >= Sd ? sp - Sd : sp);
      acc += srow[lr][sp] * w1[j];
    }
    sloc[lr][s] = acc;
  }
  __syncthreads();

  float* tr = t + (long long)b * CES + (long long)c * Sd;
#pragma unroll
  for (int p = 0; p < 3; ++p) {
    const int s = lane + p * 32;
    float acc = sloc[lr][s] + b2;            // t = local + dilated(local) + db2
#pragma unroll
    for (int j = 0; j < 11; ++j) {
      int sp = s + 5 * (j - 5);
      sp = (sp < 0) ? sp + Sd : (sp >= Sd ? sp - Sd : sp);
      acc += sloc[lr][sp] * w2[j];
    }
    tr[s] = acc;
  }
}

// ---------------------------------------------------------------------------
// Kernel 2: IVGC — fused 3-stage block-diagonal GEMM via WMMA bf16.
// grid = (24 groups, 16 batch-chunks); block = 256 = 8 waves (one M-tile each).
// Per (b,g): u_blk = Wo_g @ (Ws_g @ Xt + Wh_g @ Xp + bs + bh) + bo.
// Ws/Wh fragments live in VGPRs (used 2x per k-step); Wo lives in an LDS
// A-panel staged once per WG (keeps the wave under the 256-VGPR window).
// X/Xp/Y panels live in LDS pre-swizzled: every fragment access = 32B/lane.
// ---------------------------------------------------------------------------
__global__ __launch_bounds__(256)
void ivgc_kernel(const float* __restrict__ t,
                 const float* __restrict__ Ws, const float* __restrict__ bs,
                 const float* __restrict__ Wh, const float* __restrict__ bh,
                 const float* __restrict__ Wo, const float* __restrict__ bo,
                 float* __restrict__ u) {
  __shared__ alignas(32) __bf16 sX  [BLK];       // 4 nt x 4 kc x 32 x 16
  __shared__ alignas(32) __bf16 sXp [BLK];
  __shared__ alignas(32) __bf16 sY  [BLK];
  __shared__ alignas(32) __bf16 sWoP[CH * CH];   // A-panel: 8 mt x 4 kc x 32 x 16
  __shared__ float sBsh[CH];
  __shared__ float sBo [CH];

  const int g     = blockIdx.x;        // 0..23
  const int b0    = blockIdx.y * 8;    // batch chunk base
  const int tid   = threadIdx.x;
  const int lane  = tid & 31;
  const int wtile = tid >> 5;          // M-tile 0..7

  if (tid < CH) {
    sBsh[tid] = bs[g * CH + tid] + bh[g * CH + tid];
    sBo[tid]  = bo[g * CH + tid];
  }

  // Stage Wo into an LDS A-panel once per WG (b128 store per 8-run).
  {
    const float* Wog = Wo + (long long)g * CH * CH;
#pragma unroll
    for (int it = 0; it < 8; ++it) {
      const int slot = it * 256 + tid;   // 0..2047
      const int row  = slot >> 4;        // 0..127
      const int kr   = (slot & 15) * 8;  // 0..120 step 8
      const float* src = Wog + row * CH + kr;
      v8bf pk;
#pragma unroll
      for (int j = 0; j < 8; ++j) pk[j] = (__bf16)src[j];
      const int klo = kr & 31;
      const int fl  = (row & 15) + 16 * ((klo >> 4) & 1);
      const int h0  = klo & 15;
      *(v8bf*)&sWoP[(((row >> 4) * 4 + (kr >> 5)) * 32 + fl) * 16 + h0] = pk;
    }
  }

  // Hoist Ws/Wh fragments (read once per WG, stay in VGPRs; 64 VGPRs total).
  v16bf As[4], Ah[4];
#pragma unroll
  for (int k = 0; k < 4; ++k) {
    As[k] = frag_A_gl(Ws + (long long)g * CH * CH, CH, wtile, k);
    Ah[k] = frag_A_gl(Wh + (long long)g * CH * CH, CH, wtile, k);
  }

  for (int bi = 0; bi < 8; ++bi) {
    const int b = b0 + bi;
    const float* tb = t + (long long)b * CES;

    // Stage Xt (contiguous chunk of flattened t) and Xp (shift/pad gather),
    // writing directly in swizzled B-panel layout: one b128 store per 8-run.
#pragma unroll
    for (int it = 0; it < 4; ++it) {
      const int slot = it * 256 + tid;     // 0..1023
      const int l    = slot & 63;          // column
      const int i0   = (slot >> 6) * 8;    // K-run base (0..120 step 8)
      v8bf xt, xp;
#pragma unroll
      for (int j = 0; j < 8; ++j) {
        const int i = i0 + j;
        xt[j] = (__bf16)tb[g * BLK + i * 64 + l];
        const int f  = g * BLK + i * 64 + l;   // flat idx in padded view
        const int v  = f / 6400;               // E*(S+4) = 6400
        const int r  = f - v * 6400;
        const int e  = r / 100;                // S+4 = 100
        const int sp = r - e * 100;
        xp[j] = (sp < 2 || sp >= 98) ? (__bf16)0.0f
                                     : (__bf16)tb[v * 6144 + e * 96 + (sp - 2)];
      }
      const int nt   = l >> 4;
      const int dl   = (l & 15) + 16 * ((i0 >> 4) & 1);
      const int kc   = i0 >> 5;
      const int addr = ((nt * 4 + kc) * 32 + dl) * 16 + (i0 & 15);
      *(v8bf*)&sX [addr] = xt;
      *(v8bf*)&sXp[addr] = xp;
    }
    if (bi + 1 < 8)
      __builtin_prefetch(t + (long long)(b + 1) * CES + g * BLK + tid, 0, 1);
    __syncthreads();

    // Stage A+B: y = Ws@Xt + Wh@Xp + (bs+bh) -> swizzled LDS panel (b128 store)
#pragma unroll
    for (int n = 0; n < 4; ++n) {
      v8f acc = {};
#pragma unroll
      for (int k = 0; k < 4; ++k) {
        acc = WMMA_BF16(As[k], frag_B_sw(sX,  4, n, k), acc);
        acc = WMMA_BF16(Ah[k], frag_B_sw(sXp, 4, n, k), acc);
      }
      const int row0 = wtile * 16 + ((lane >> 4) << 3);  // 8-aligned
      v8bf yv;
#pragma unroll
      for (int r8 = 0; r8 < 8; ++r8)
        yv[r8] = (__bf16)(acc[r8] + sBsh[row0 + r8]);
      const int dl   = (lane & 15) + 16 * ((row0 >> 4) & 1);
      const int addr = ((n * 4 + (row0 >> 5)) * 32 + dl) * 16 + (row0 & 15);
      *(v8bf*)&sY[addr] = yv;
    }
    __syncthreads();

    // Stage C: u = Wo@y + bo -> global f32 (Wo fragments from LDS A-panel)
    float* ub = u + (long long)b * CES + g * BLK;
#pragma unroll
    for (int n = 0; n < 4; ++n) {
      v8f acc = {};
#pragma unroll
      for (int k = 0; k < 4; ++k)
        acc = WMMA_BF16(frag_A_sw(sWoP, 4, wtile, k), frag_B_sw(sY, 4, n, k), acc);
      const int col  = n * 16 + (lane & 15);
      const int row0 = wtile * 16 + ((lane >> 4) << 3);
#pragma unroll
      for (int r8 = 0; r8 < 8; ++r8)
        ub[(row0 + r8) * 64 + col] = acc[r8] + sBo[row0 + r8];
    }
    __syncthreads();   // protect sX/sXp/sY before next batch
  }
}

// ---------------------------------------------------------------------------
// Kernel 3: pooled means (contiguous reductions; float4 vectorized)
// ---------------------------------------------------------------------------
__global__ void pool_t_kernel(const float* __restrict__ u, float* __restrict__ pt) {
  const int idx = blockIdx.x * 256 + threadIdx.x;   // 128*6144
  const int b = idx / 6144, j = idx - b * 6144;
  const float4* p = (const float4*)(u + (long long)b * CES + j * 32);
  float s = 0.f;
#pragma unroll
  for (int k = 0; k < 8; ++k) { float4 q = p[k]; s += q.x + q.y + q.z + q.w; }
  pt[idx] = s * (1.0f / 32.0f);
}

__global__ void pool_v_kernel(const float* __restrict__ u, float* __restrict__ pv) {
  const int idx = blockIdx.x * 256 + threadIdx.x;   // 128*2048
  const int b = idx / 2048, j = idx - b * 2048;
  const float4* p = (const float4*)(u + (long long)b * CES + j * 96);
  float s = 0.f;
#pragma unroll
  for (int k = 0; k < 24; ++k) { float4 q = p[k]; s += q.x + q.y + q.z + q.w; }
  pv[idx] = s * (1.0f / 96.0f);
}

// ---------------------------------------------------------------------------
// Kernel 4: generic WMMA GEMM  out[M=128,N] = act(A[128,K] @ W[N,K]^T + b[N])
// block = 256 (8 waves -> 8 M-tiles); grid.x = N/16; act: 0=relu, 1=sigmoid.
// A (128x32) and B (32x16) k-slabs staged in LDS in fragment layout per chunk.
// ---------------------------------------------------------------------------
__global__ __launch_bounds__(256)
void gemm_bias_act_kernel(const float* __restrict__ A, const float* __restrict__ W,
                          const float* __restrict__ bias, float* __restrict__ out,
                          int N, int K, int act) {
  __shared__ alignas(32) __bf16 sA[8 * 32 * 16];   // 8 mtiles x 32 lanes x 16
  __shared__ alignas(32) __bf16 sB[32 * 16];       // 1 ntile  x 32 lanes x 16
  const int tid   = threadIdx.x;
  const int lane  = tid & 31;
  const int wtile = tid >> 5;
  const int colb  = blockIdx.x * 16;
  v8f acc = {};
  const int nk = K >> 5;
  for (int kc = 0; kc < nk; ++kc) {
    const int kbase = kc * 32;
    // stage A: 128 rows x 32 k = 512 runs of 8 -> 2 per thread
#pragma unroll
    for (int it = 0; it < 2; ++it) {
      const int slot = it * 256 + tid;     // 0..511
      const int row  = slot >> 2;          // 0..127
      const int kr   = (slot & 3) * 8;     // 0,8,16,24
      const float* src = A + (long long)row * K + kbase + kr;
      v8bf pk;
#pragma unroll
      for (int j = 0; j < 8; ++j) pk[j] = (__bf16)src[j];
      const int dl = (row & 15) + 16 * ((kr >> 4) & 1);
      const int h0 = kr & 15;
      *(v8bf*)&sA[((row >> 4) * 32 + dl) * 16 + h0] = pk;
    }
    // stage B: 16 cols x 32 k = 64 runs of 8 -> threads 0..63
    if (tid < 64) {
      const int col = tid >> 2;
      const int kr  = (tid & 3) * 8;
      const float* src = W + (long long)(colb + col) * K + kbase + kr;
      v8bf pk;
#pragma unroll
      for (int j = 0; j < 8; ++j) pk[j] = (__bf16)src[j];
      *(v8bf*)&sB[(col + 16 * (kr >> 4)) * 16 + (kr & 15)] = pk;
    }
    __syncthreads();
    v16bf a = *(const v16bf*)&sA[(wtile * 32 + lane) * 16];
    v16bf b = *(const v16bf*)&sB[lane * 16];
    acc = WMMA_BF16(a, b, acc);
    __syncthreads();
  }
  const int col = colb + (lane & 15);
  const float bv = bias[col];
#pragma unroll
  for (int r8 = 0; r8 < 8; ++r8) {
    const int row = wtile * 16 + r8 + ((lane >> 4) << 3);
    float v = acc[r8] + bv;
    v = (act == 0) ? fmaxf(v, 0.0f) : 1.0f / (1.0f + __expf(-v));
    out[(long long)row * N + col] = v;
  }
}

// ---------------------------------------------------------------------------
// Kernel 5: final gating  out = sigmoid(ta * va * u) * x   (4 elems/thread)
// 96 % 4 == 0 and 6144 % 4 == 0 -> a 4-run never crosses an e or v boundary.
// ---------------------------------------------------------------------------
__global__ void gate_kernel(const float* __restrict__ x, const float* __restrict__ u,
                            const float* __restrict__ ta, const float* __restrict__ va,
                            float* __restrict__ out) {
  const long long idx = ((long long)blockIdx.x * 256 + threadIdx.x) * 4;
  const int b  = (int)(idx / CES);
  const int r  = (int)(idx - (long long)b * CES);
  const int v  = r / 6144;            // E*S
  const int r2 = r - v * 6144;        // e*96 + s
  const int e  = r2 / 96;
  const float4 t4 = *(const float4*)(ta + b * 6144 + r2);
  const float  vv = va[b * 2048 + v * 64 + e];
  const float4 u4 = *(const float4*)(u + idx);
  const float4 x4 = *(const float4*)(x + idx);
  float4 o;
  o.x = x4.x / (1.0f + __expf(-(t4.x * vv * u4.x)));
  o.y = x4.y / (1.0f + __expf(-(t4.y * vv * u4.y)));
  o.z = x4.z / (1.0f + __expf(-(t4.z * vv * u4.z)));
  o.w = x4.w / (1.0f + __expf(-(t4.w * vv * u4.w)));
  *(float4*)(out + idx) = o;
}

// ---------------------------------------------------------------------------
// Host-side launch
// ---------------------------------------------------------------------------
extern "C" void kernel_launch(void* const* d_in, const int* in_sizes, int n_in,
                              void* d_out, int out_size, void* d_ws, size_t ws_size,
                              hipStream_t stream) {
  const float* x   = (const float*)d_in[0];
  const float* dw1 = (const float*)d_in[1];
  const float* db1 = (const float*)d_in[2];
  const float* dw2 = (const float*)d_in[3];
  const float* db2 = (const float*)d_in[4];
  const float* Ws  = (const float*)d_in[5];
  const float* bs  = (const float*)d_in[6];
  const float* Wh  = (const float*)d_in[7];
  const float* bh  = (const float*)d_in[8];
  const float* Wo  = (const float*)d_in[9];
  const float* bo  = (const float*)d_in[10];
  const float* Wt1 = (const float*)d_in[11];
  const float* bt1 = (const float*)d_in[12];
  const float* Wt2 = (const float*)d_in[13];
  const float* bt2 = (const float*)d_in[14];
  const float* Wv1 = (const float*)d_in[15];
  const float* bv1 = (const float*)d_in[16];
  const float* Wv2 = (const float*)d_in[17];
  const float* bv2 = (const float*)d_in[18];

  char* ws = (char*)d_ws;
  float* t_buf = (float*)(ws + 0);              // 128*196608*4 = 100663296 B
  float* u_buf = (float*)(ws + 100663296);      // 100663296 B
  float* pt    = (float*)(ws + 201326592);      // 3145728 B
  float* pv    = (float*)(ws + 204472320);      // 1048576 B
  float* h1    = (float*)(ws + 205520896);      // 196608 B
  float* ta    = (float*)(ws + 205717504);      // 3145728 B
  float* h2    = (float*)(ws + 208863232);      // 65536 B
  float* va    = (float*)(ws + 208928768);      // 1048576 B

  // 1) TLDC: x -> t
  tldc_kernel<<<32768, 256, 0, stream>>>(x, dw1, db1, dw2, db2, t_buf);

  // 2) IVGC: t -> u, fused 3-stage WMMA block-diagonal GEMM
  ivgc_kernel<<<dim3(NG, 16), 256, 0, stream>>>(t_buf, Ws, bs, Wh, bh, Wo, bo, u_buf);

  // 3) GTVA pooling
  pool_t_kernel<<<(128 * 6144) / 256, 256, 0, stream>>>(u_buf, pt);
  pool_v_kernel<<<(128 * 2048) / 256, 256, 0, stream>>>(u_buf, pv);

  // 4) GTVA MLPs (WMMA GEMMs)
  gemm_bias_act_kernel<<<384  / 16, 256, 0, stream>>>(pt, Wt1, bt1, h1, 384, 6144, 0);
  gemm_bias_act_kernel<<<6144 / 16, 256, 0, stream>>>(h1, Wt2, bt2, ta, 6144, 384, 1);
  gemm_bias_act_kernel<<<128  / 16, 256, 0, stream>>>(pv, Wv1, bv1, h2, 128, 2048, 0);
  gemm_bias_act_kernel<<<2048 / 16, 256, 0, stream>>>(h2, Wv2, bv2, va, 2048, 128, 1);

  // 5) Final gating
  gate_kernel<<<(int)(((long long)Bn * CES) / 1024), 256, 0, stream>>>(
      x, u_buf, ta, va, (float*)d_out);
}
